// BertNer_80934363726535
// MI455X (gfx1250) — matrix operand order repaired and checked by reading, not compiled, not run on hardware
//
#include <hip/hip_runtime.h>
#include <math.h>

// ---------------------------------------------------------------------------
// BertNer pipeline for gfx1250 (MI455X). All dense GEMMs run on
// v_wmma_f32_16x16x32_f16 (wave32). GEMM k-loop is manually unrolled x2 with
// two independent fragment stage sets so every WMMA pair has the other
// stage's loads in flight (partial s_wait_loadcnt, never a full stall).
// LSTM recurrence: persistent workgroup per direction, Whh fragments resident
// in VGPRs for all 256 steps, next-step gx loads overlapped with compute.
// ---------------------------------------------------------------------------

typedef __attribute__((ext_vector_type(16))) _Float16 v16h;
typedef __attribute__((ext_vector_type(8)))  _Float16 v8h;
typedef __attribute__((ext_vector_type(8)))  float    v8f;

#define DEV __device__ __forceinline__

DEV float sigf(float x) { return 1.0f / (1.0f + expf(-x)); }

DEV v16h pack16(v8h lo, v8h hi) {
    v16h r;
#pragma unroll
    for (int i = 0; i < 8; ++i) { r[i] = lo[i]; r[8 + i] = hi[i]; }
    return r;
}

// A fragment: 16(M) x 32(K) fp16 tile from row-major A (leading dim lda halves).
// ISA layout: lane<16 -> M=lane, K in {0..7, 16..23}; lane>=16 -> K in {8..15, 24..31}.
DEV v16h load_a_frag(const _Float16* A, int lda, int m0, int k0, int lane) {
    int hf = lane >> 4, ls = lane & 15;
    const _Float16* p = A + (size_t)(m0 + ls) * lda + k0 + 8 * hf;
    v8h lo = *reinterpret_cast<const v8h*>(p);
    v8h hi = *reinterpret_cast<const v8h*>(p + 16);
    return pack16(lo, hi);
}

// B fragment: 32(K) x 16(N); W stored [N][K] row-major, B[k][n] = W[n0+n][k0+k].
// ISA layout: lane<16 -> N=lane, K in {0..7, 16..23}; lane>=16 -> K in {8..15, 24..31}.
DEV v16h load_b_frag(const _Float16* W, int ldw, int n0, int k0, int lane) {
    int hf = lane >> 4, ls = lane & 15;
    const _Float16* p = W + (size_t)(n0 + ls) * ldw + k0 + 8 * hf;
    v8h lo = *reinterpret_cast<const v8h*>(p);
    v8h hi = *reinterpret_cast<const v8h*>(p + 16);
    return pack16(lo, hi);
}

DEV v8f wmma16(v16h a, v16h b, v8f c) {
    return __builtin_amdgcn_wmma_f32_16x16x32_f16(false, a, false, b, (short)0, c,
                                                  false, false);
}

// ----------------------------- packing kernels -----------------------------

__global__ void k_cvt_f16(const float* __restrict__ s, _Float16* __restrict__ d, int n) {
    int i = blockIdx.x * 256 + threadIdx.x;
    if (i < n) d[i] = (_Float16)s[i];
}

// dst[c * srcRows + r] = src[r * srcCols + c]   (fp32 -> fp16 transpose)
__global__ void k_transpose_f16(const float* __restrict__ s, _Float16* __restrict__ d,
                                int srcRows, int srcCols) {
    int i = blockIdx.x * 256 + threadIdx.x;
    if (i < srcRows * srcCols) {
        int c = i / srcRows;
        int r = i % srcRows;
        d[i] = (_Float16)s[(size_t)r * srcCols + c];
    }
}

__global__ void k_biascat(const float* __restrict__ bif, const float* __restrict__ bhf,
                          const float* __restrict__ bib, const float* __restrict__ bhb,
                          float* __restrict__ out) {
    int i = blockIdx.x * 256 + threadIdx.x;
    if (i < 512)        out[i] = bif[i] + bhf[i];
    else if (i < 1024)  out[i] = bib[i - 512] + bhb[i - 512];
}

// node_emb = mean over TN=3 of emb_table rows; 4800 rows of 768, output fp16.
__global__ void k_node_emb(const float* __restrict__ emb, const int* __restrict__ ids,
                           _Float16* __restrict__ out) {
    int row = blockIdx.x;
    int i0 = ids[row * 3 + 0], i1 = ids[row * 3 + 1], i2 = ids[row * 3 + 2];
    const float* e0 = emb + (size_t)i0 * 768;
    const float* e1 = emb + (size_t)i1 * 768;
    const float* e2 = emb + (size_t)i2 * 768;
    for (int h = threadIdx.x; h < 768; h += 256)
        out[(size_t)row * 768 + h] = (_Float16)((e0[h] + e1[h] + e2[h]) * (1.0f / 3.0f));
}

// ------------------------------- WMMA GEMM ---------------------------------
// C[M][N] = A[M][K](fp16) * W[N][K]^T(fp16) + bias. 256 thr = 8 waves (4Mx2N),
// each wave: one 16x32 C tile (two 16x16 WMMA accumulators).
// Requirements (hold at every call site): M%64==0, N%32==0, K%64==0.
// Both 16-wide tiles of an active wave are always valid -> no divergent
// branches, EXEC all-ones at every WMMA. K-loop: manual unroll x2, two
// independent stage sets (A/B) so loads of one stage fly behind the other
// stage's WMMAs.

__global__ __launch_bounds__(256)
void k_wmma_gemm(const _Float16* __restrict__ A, const _Float16* __restrict__ W,
                 const float* __restrict__ bias, float* __restrict__ Cf,
                 _Float16* __restrict__ Ch, int M, int N, int K) {
    int lane = threadIdx.x & 31;
    int w    = __builtin_amdgcn_readfirstlane(threadIdx.x >> 5);   // scalar wave id
    int wm   = w & 3, wn = w >> 2;
    int m0 = blockIdx.x * 64 + wm * 16;
    int n0 = blockIdx.y * 64 + wn * 32;
    if (m0 >= M || n0 >= N) return;          // scalar branch, EXEC untouched
    int n1 = n0 + 16;

    v8f acc0, acc1;
#pragma unroll
    for (int r = 0; r < 8; ++r) { acc0[r] = 0.0f; acc1[r] = 0.0f; }

    // prologue: two pipeline stages (k = 0 and k = 32)
    v16h aA  = load_a_frag(A, K, m0, 0, lane);
    v16h b0A = load_b_frag(W, K, n0, 0, lane);
    v16h b1A = load_b_frag(W, K, n1, 0, lane);
    v16h aB  = load_a_frag(A, K, m0, 32, lane);
    v16h b0B = load_b_frag(W, K, n0, 32, lane);
    v16h b1B = load_b_frag(W, K, n1, 32, lane);

    int k0 = 0;
    for (; k0 + 64 < K; k0 += 64) {
        // refill stage A (k0+64) while stage-A WMMAs consume old registers
        v16h aA2  = load_a_frag(A, K, m0, k0 + 64, lane);
        v16h c0A  = load_b_frag(W, K, n0, k0 + 64, lane);
        v16h c1A  = load_b_frag(W, K, n1, k0 + 64, lane);
        acc0 = wmma16(aA, b0A, acc0);
        acc1 = wmma16(aA, b1A, acc1);
        // refill stage B (k0+96) while stage-B WMMAs run
        v16h aB2  = load_a_frag(A, K, m0, k0 + 96, lane);
        v16h c0B  = load_b_frag(W, K, n0, k0 + 96, lane);
        v16h c1B  = load_b_frag(W, K, n1, k0 + 96, lane);
        acc0 = wmma16(aB, b0B, acc0);
        acc1 = wmma16(aB, b1B, acc1);
        aA = aA2; b0A = c0A; b1A = c1A;
        aB = aB2; b0B = c0B; b1B = c1B;
        if (k0 + 128 < K)
            __builtin_prefetch(A + (size_t)(m0 + (lane & 15)) * K + k0 + 128, 0, 3);
    }
    // drain both stages
    acc0 = wmma16(aA, b0A, acc0);
    acc1 = wmma16(aA, b1A, acc1);
    acc0 = wmma16(aB, b0B, acc0);
    acc1 = wmma16(aB, b1B, acc1);

    int hf = lane >> 4, ls = lane & 15;
#pragma unroll
    for (int r = 0; r < 8; ++r) {
        int m = m0 + r + 8 * hf;
        int n = n0 + ls;
        float v = acc0[r] + (bias ? bias[n] : 0.0f);
        if (Cf) Cf[(size_t)m * N + n] = v;
        if (Ch) Ch[(size_t)m * N + n] = (_Float16)v;
    }
#pragma unroll
    for (int r = 0; r < 8; ++r) {
        int m = m0 + r + 8 * hf;
        int n = n1 + ls;
        float v = acc1[r] + (bias ? bias[n] : 0.0f);
        if (Cf) Cf[(size_t)m * N + n] = v;
        if (Ch) Ch[(size_t)m * N + n] = (_Float16)v;
    }
}

// ------------------------------ GCN epilogue -------------------------------
// Only row n=0 of gcn_out is consumed: sememe[lws][h] = relu(adj[lws][0][:] .
// support[lws][:][h] + gcn_b[h]).  400 blocks x 256 threads.
__global__ void k_sememe(const float* __restrict__ support, const float* __restrict__ adj,
                         const float* __restrict__ gcn_b, float* __restrict__ sem) {
    int row = blockIdx.x;                          // (l*4+w)*5+s
    const float* arow = adj + (size_t)row * 144;   // adj[l][w][s][0][m], N=12
    float a[12];
#pragma unroll
    for (int m = 0; m < 12; ++m) a[m] = arow[m];
    for (int h = threadIdx.x; h < 768; h += 256) {
        float acc = gcn_b[h];
#pragma unroll
        for (int m = 0; m < 12; ++m)
            acc += a[m] * support[((size_t)row * 12 + m) * 768 + h];
        sem[(size_t)row * 768 + h] = fmaxf(acc, 0.0f);
    }
}

// dist -> softmax over S=5 -> att_mean -> word_rep. One block per (l,w).
__global__ void k_wordrep(const float* __restrict__ emb, const int* __restrict__ word_ids,
                          const float* __restrict__ sem, float* __restrict__ wrep) {
    __shared__ float red[256];
    __shared__ float sdist[5];
    __shared__ float sattn[5];
    int blk = blockIdx.x, tid = threadIdx.x;
    const float* svec = emb + (size_t)word_ids[blk] * 768;
    const float* semb = sem + (size_t)blk * 5 * 768;
    for (int s = 0; s < 5; ++s) {
        float p = 0.0f;
        for (int h = tid; h < 768; h += 256) {
            float d = svec[h] - semb[s * 768 + h] + 1e-6f;
            p += d * d;
        }
        red[tid] = p; __syncthreads();
        for (int st = 128; st > 0; st >>= 1) {
            if (tid < st) red[tid] += red[tid + st];
            __syncthreads();
        }
        if (tid == 0) sdist[s] = sqrtf(red[0]);
        __syncthreads();
    }
    if (tid == 0) {
        float mx = sdist[0];
        for (int s = 1; s < 5; ++s) mx = fmaxf(mx, sdist[s]);
        float sum = 0.0f;
        for (int s = 0; s < 5; ++s) { float e = expf(sdist[s] - mx); sattn[s] = e; sum += e; }
        for (int s = 0; s < 5; ++s) sattn[s] /= sum;
    }
    __syncthreads();
    for (int h = tid; h < 768; h += 256) {
        float am = 0.0f;
        for (int s = 0; s < 5; ++s) am += sattn[s] * semb[s * 768 + h];
        am *= 0.2f;                                   // mean over S
        wrep[(size_t)blk * 768 + h] = 0.5f * (am + svec[h]);
    }
}

// label_rep = mean over W=4; pad to 32 rows (rows 20..31 = 0) for N=32 GEMM.
__global__ void k_labelrep(const float* __restrict__ wrep, _Float16* __restrict__ lrepH) {
    int l = blockIdx.x;
    for (int h = threadIdx.x; h < 768; h += 256) {
        float v = 0.0f;
        if (l < 20)
            v = 0.25f * (wrep[((size_t)l * 4 + 0) * 768 + h] +
                         wrep[((size_t)l * 4 + 1) * 768 + h] +
                         wrep[((size_t)l * 4 + 2) * 768 + h] +
                         wrep[((size_t)l * 4 + 3) * 768 + h]);
        lrepH[(size_t)l * 768 + h] = (_Float16)v;
    }
}

// --------------------------- LSTM recurrence -------------------------------
// 2 blocks (dir 0 = fwd, dir 1 = bwd) x 1024 threads (32 waves). Wave w owns
// gate units [16w, 16w+16). Whh B-fragments (K=128 -> 4 tiles) live in VGPRs
// for all 256 steps. h (fp16) / c (fp32) / gates (fp32) staged in LDS.
// Next-step gx loads are issued before this step's WMMAs so their latency
// hides behind compute + the two barriers.
__global__ __launch_bounds__(1024)
void k_lstm(const float* __restrict__ gx, const _Float16* __restrict__ WhhH,
            _Float16* __restrict__ seqH) {
    const int dir = blockIdx.x;
    __shared__ float sGates[16 * 512];                 // 32 KB
    __shared__ float sC[16 * 128];                     //  8 KB
    __shared__ __align__(16) _Float16 sH[16 * 128];    //  4 KB
    int tid = threadIdx.x;
    int lane = tid & 31;
    int w = __builtin_amdgcn_readfirstlane(tid >> 5);  // scalar wave id
    int hf = lane >> 4, ls = lane & 15;
    int u0 = w * 16;

    for (int i = tid; i < 16 * 128; i += 1024) { sC[i] = 0.0f; sH[i] = (_Float16)0.0f; }
    __syncthreads();

    const _Float16* Wd = WhhH + (size_t)dir * 512 * 128;
    v16h bfrag[4];
#pragma unroll
    for (int kt = 0; kt < 4; ++kt) bfrag[kt] = load_b_frag(Wd, 128, u0, kt * 32, lane);

    // prologue: gx for step 0
    v8f gxr;
    {
        int t0 = dir ? 255 : 0;
#pragma unroll
        for (int r = 0; r < 8; ++r) {
            int b = r + 8 * hf;
            gxr[r] = gx[((size_t)b * 256 + t0) * 1024 + dir * 512 + u0 + ls];
        }
    }

    for (int step = 0; step < 256; ++step) {
        int t = dir ? (255 - step) : step;

        // issue next step's gx loads first; they complete during this step
        v8f gxn;
        if (step + 1 < 256) {
            int tn = dir ? (254 - step) : (step + 1);
#pragma unroll
            for (int r = 0; r < 8; ++r) {
                int b = r + 8 * hf;
                gxn[r] = gx[((size_t)b * 256 + tn) * 1024 + dir * 512 + u0 + ls];
            }
        } else {
#pragma unroll
            for (int r = 0; r < 8; ++r) gxn[r] = 0.0f;
        }

        v8f acc = gxr;                                 // bias already folded in
#pragma unroll
        for (int kt = 0; kt < 4; ++kt) {
            v16h a = load_a_frag((const _Float16*)sH, 128, 0, kt * 32, lane);
            acc = wmma16(a, bfrag[kt], acc);
        }
#pragma unroll
        for (int r = 0; r < 8; ++r) {
            int b = r + 8 * hf;
            sGates[b * 512 + u0 + ls] = acc[r];
        }
        __syncthreads();

#pragma unroll
        for (int rep = 0; rep < 2; ++rep) {
            int idx = tid + rep * 1024;                // 2048 (b,hu) pairs
            int b = idx >> 7, hu = idx & 127;
            float gi = sGates[b * 512 +       hu];
            float gf = sGates[b * 512 + 128 + hu];
            float gg = sGates[b * 512 + 256 + hu];
            float go = sGates[b * 512 + 384 + hu];
            float c = sigf(gf) * sC[idx] + sigf(gi) * tanhf(gg);
            float h = sigf(go) * tanhf(c);
            sC[idx] = c;
            _Float16 hh = (_Float16)h;
            sH[idx] = hh;                              // idx == b*128 + hu
            seqH[((size_t)b * 256 + t) * 256 + dir * 128 + hu] = hh;
        }
        __syncthreads();

        gxr = gxn;
    }
}

// -------------------------- softmax / loss ---------------------------------
// One wave per (b,t) row: softmax over L=20 (cols 20..31 are padding).
__global__ __launch_bounds__(256)
void k_softmax(const float* __restrict__ scores, const int* __restrict__ labels,
               float* __restrict__ probs, float* __restrict__ nll) {
    int tid = threadIdx.x, lane = tid & 31;
    int row = blockIdx.x * 8 + (tid >> 5);
    const float* sr = scores + (size_t)row * 32;
    float x = (lane < 20) ? sr[lane] : -1e30f;
    float m = x;
    for (int off = 16; off > 0; off >>= 1) m = fmaxf(m, __shfl_xor(m, off, 32));
    float e = expf(x - m);
    float s = e;
    for (int off = 16; off > 0; off >>= 1) s += __shfl_xor(s, off, 32);
    if (lane < 20) probs[(size_t)row * 20 + lane] = e / s;
    if (lane == 0) {
        int lab = labels[row];
        nll[row] = (m + logf(s)) - sr[lab];
    }
}

__global__ void k_loss(const float* __restrict__ nll, float* __restrict__ out) {
    __shared__ float red[256];
    int tid = threadIdx.x;
    float p = 0.0f;
    for (int i = tid; i < 4096; i += 256) p += nll[i];
    red[tid] = p; __syncthreads();
    for (int st = 128; st > 0; st >>= 1) {
        if (tid < st) red[tid] += red[tid + st];
        __syncthreads();
    }
    if (tid == 0) out[0] = red[0] * (1.0f / 4096.0f);
}

// ------------------------------- launcher ----------------------------------

extern "C" void kernel_launch(void* const* d_in, const int* in_sizes, int n_in,
                              void* d_out, int out_size, void* d_ws, size_t ws_size,
                              hipStream_t stream) {
    const float* tokemb  = (const float*)d_in[0];
    const float* emb     = (const float*)d_in[1];
    const float* gcn_w   = (const float*)d_in[2];
    const float* gcn_b   = (const float*)d_in[3];
    const float* Wih_f   = (const float*)d_in[4];
    const float* Whh_f   = (const float*)d_in[5];
    const float* bih_f   = (const float*)d_in[6];
    const float* bhh_f   = (const float*)d_in[7];
    const float* Wih_b   = (const float*)d_in[8];
    const float* Whh_b   = (const float*)d_in[9];
    const float* bih_b   = (const float*)d_in[10];
    const float* bhh_b   = (const float*)d_in[11];
    const float* lin_w   = (const float*)d_in[12];
    const float* lin_b   = (const float*)d_in[13];
    const float* adj     = (const float*)d_in[14];
    const int*   word_ids = (const int*)d_in[15];
    const int*   node_ids = (const int*)d_in[16];
    const int*   labels   = (const int*)d_in[17];
    float* out = (float*)d_out;
    (void)in_sizes; (void)n_in; (void)out_size; (void)ws_size;

    char* ws = (char*)d_ws;
    size_t off = 0;
    auto alloc = [&](size_t bytes) -> char* {
        char* p = ws + off;
        off += (bytes + 255) & ~(size_t)255;
        return p;
    };

    _Float16* nodeEmbH = (_Float16*)alloc((size_t)4800 * 768 * 2);
    _Float16* gcnWT    = (_Float16*)alloc((size_t)768 * 768 * 2);
    float*    support  = (float*)   alloc((size_t)4800 * 768 * 4);
    float*    sem      = (float*)   alloc((size_t)400 * 768 * 4);
    float*    wrep     = (float*)   alloc((size_t)80 * 768 * 4);
    _Float16* lrepH    = (_Float16*)alloc((size_t)32 * 768 * 2);
    _Float16* WihCatH  = (_Float16*)alloc((size_t)1024 * 768 * 2);
    float*    biasCat  = (float*)   alloc((size_t)1024 * 4);
    _Float16* WhhCatH  = (_Float16*)alloc((size_t)2 * 512 * 128 * 2);
    _Float16* linWH    = (_Float16*)alloc((size_t)768 * 256 * 2);
    _Float16* tokEmbH  = (_Float16*)alloc((size_t)4096 * 768 * 2);
    float*    gx       = (float*)   alloc((size_t)4096 * 1024 * 4);
    _Float16* seqH     = (_Float16*)alloc((size_t)4096 * 256 * 2);
    _Float16* tokH     = (_Float16*)alloc((size_t)4096 * 768 * 2);
    float*    scores   = (float*)   alloc((size_t)4096 * 32 * 4);
    float*    nll      = (float*)   alloc((size_t)4096 * 4);

    // ---- pack / convert ----
    k_node_emb<<<4800, 256, 0, stream>>>(emb, node_ids, nodeEmbH);
    k_transpose_f16<<<(768 * 768 + 255) / 256, 256, 0, stream>>>(gcn_w, gcnWT, 768, 768);
    k_cvt_f16<<<(512 * 768 + 255) / 256, 256, 0, stream>>>(Wih_f, WihCatH, 512 * 768);
    k_cvt_f16<<<(512 * 768 + 255) / 256, 256, 0, stream>>>(Wih_b, WihCatH + (size_t)512 * 768, 512 * 768);
    k_biascat<<<4, 256, 0, stream>>>(bih_f, bhh_f, bih_b, bhh_b, biasCat);
    k_cvt_f16<<<(512 * 128 + 255) / 256, 256, 0, stream>>>(Whh_f, WhhCatH, 512 * 128);
    k_cvt_f16<<<(512 * 128 + 255) / 256, 256, 0, stream>>>(Whh_b, WhhCatH + (size_t)512 * 128, 512 * 128);
    k_cvt_f16<<<(768 * 256 + 255) / 256, 256, 0, stream>>>(lin_w, linWH, 768 * 256);
    k_cvt_f16<<<(4096 * 768 + 255) / 256, 256, 0, stream>>>(tokemb, tokEmbH, 4096 * 768);

    // ---- GCN / label branch ----
    k_wmma_gemm<<<dim3(4800 / 64, 768 / 64), 256, 0, stream>>>(
        nodeEmbH, gcnWT, nullptr, support, nullptr, 4800, 768, 768);
    k_sememe<<<400, 256, 0, stream>>>(support, adj, gcn_b, sem);
    k_wordrep<<<80, 256, 0, stream>>>(emb, word_ids, sem, wrep);
    k_labelrep<<<32, 256, 0, stream>>>(wrep, lrepH);

    // ---- BiLSTM ----
    k_wmma_gemm<<<dim3(4096 / 64, 1024 / 64), 256, 0, stream>>>(
        tokEmbH, WihCatH, biasCat, gx, nullptr, 4096, 1024, 768);
    k_lstm<<<2, 1024, 0, stream>>>(gx, WhhCatH, seqH);

    // ---- projection + scores ----
    k_wmma_gemm<<<dim3(4096 / 64, 768 / 64), 256, 0, stream>>>(
        seqH, linWH, lin_b, nullptr, tokH, 4096, 768, 256);
    k_wmma_gemm<<<dim3(4096 / 64, 1), 256, 0, stream>>>(
        tokH, lrepH, nullptr, scores, nullptr, 4096, 32, 768);

    // ---- softmax / loss ----
    k_softmax<<<512, 256, 0, stream>>>(scores, labels, out, nll);
    k_loss<<<1, 256, 0, stream>>>(nll, out + 81920);
}
